// neural_network_48653389529231
// MI455X (gfx1250) — matrix operand order repaired and checked by reading, not compiled, not run on hardware
//
#include <hip/hip_runtime.h>

// ---------------------------------------------------------------------------
// CDNA5 (gfx1250) implementation of the cohesive/plasticity RNN surrogate.
// Pipeline:
//   K0 prep   : fold plane-stress D into W12, convert weights to f16 (pad W2->16 rows)
//   K1 x2h    : x f32 -> f16 (reused by GEMM1 and GEMM2 A-operands)
//   K2 gemm1  : jumps = lrelu(x @ W11^T + b11)  [WMMA, B panel via TDM], fused d_new
//   K3 cummax : dmg[b,t,c] = running max_t d_new  (16384 parallel chains)
//   K4 gemm2  : sig_tr = [x | dmg] @ (D*W12)^T   [WMMA, 2 M-tiles/wave, TDM panel]
//   K5 plast  : per-(b,p) radial-return recurrence over T, writes f16 sig
//   K6 gemm3  : out = softplus(sig @ W2^T)       [WMMA, N padded 6->16, TDM panel]
// ---------------------------------------------------------------------------

typedef __attribute__((ext_vector_type(16))) _Float16 v16h;
typedef __attribute__((ext_vector_type(8)))  float    v8f;
typedef __attribute__((ext_vector_type(4)))  _Float16 v4h;
typedef __attribute__((ext_vector_type(4)))  unsigned int v4u;
typedef __attribute__((ext_vector_type(8)))  int      v8i;
typedef __attribute__((ext_vector_type(4)))  int      v4i;

#if defined(__has_builtin)
#if __has_builtin(__builtin_amdgcn_tensor_load_to_lds) && \
    __has_builtin(__builtin_amdgcn_s_wait_tensorcnt)
#define NN48_HAVE_TDM 1
#endif
#endif
#ifndef NN48_HAVE_TDM
#define NN48_HAVE_TDM 0
#endif

namespace cfg {
constexpr int B = 256, T = 512, F = 128;
constexpr int COH = 64;                 // coh_pts
constexpr int COH_SIZE = 128;           // 2*COH
constexpr int IN_BULK = 192;            // F + COH
constexpr int BULK = 128;               // bulk_pts
constexpr int BULK_SIZE = 384;          // 3*BULK
constexpr int OUT = 6;
constexpr int BT = B * T;               // 131072 rows, 8192 m-tiles of 16
constexpr float LEAKY = 0.01f;
constexpr float DELTA0 = 0.01f, DELTAF = 0.1f;
constexpr float E_MOD = 1.0e3f, NU = 0.3f;
constexpr float G_MOD = E_MOD / (2.0f * (1.0f + NU));
constexpr float SIG_Y = 10.0f, HARD = 100.0f;
constexpr float EPS = 1e-12f;
}

// ---- workspace layout (bytes) ---------------------------------------------
namespace wsl {
constexpr size_t o_w11h  = 0;                                   // 128*128 f16
constexpr size_t o_w12ph = o_w11h  + 128ull * 128 * 2;          // 384*192 f16
constexpr size_t o_w2h   = o_w12ph + 384ull * 192 * 2;          // 16*384  f16
constexpr size_t o_xh    = (o_w2h + 16ull * 384 * 2 + 255) & ~255ull;
constexpr size_t o_dnew  = o_xh   + (size_t)cfg::BT * cfg::F   * 2;  // f32
constexpr size_t o_dmgh  = o_dnew + (size_t)cfg::BT * cfg::COH * 4;  // f16
constexpr size_t o_sig   = (o_dmgh + (size_t)cfg::BT * cfg::COH * 2 + 255) & ~255ull; // f32
constexpr size_t o_sigh  = o_sig  + (size_t)cfg::BT * cfg::BULK_SIZE * 4;             // f16
}

// ---- WMMA helpers ----------------------------------------------------------

__device__ __forceinline__ v8f wmma16(v16h a, v16h b, v8f c) {
  // D = A(16x32 f16) * B(32x16 f16) + C(16x16 f32)
  return __builtin_amdgcn_wmma_f32_16x16x32_f16(
      /*neg_a=*/false, a, /*neg_b=*/false, b,
      /*c_mod=*/(short)0, c, /*reuse_a=*/false, /*reuse_b=*/false);
}

// A fragment, row-major source. ISA 7.12.2 (16-bit A 16x32):
// lanes 0-15: M=lane,   halves 0-7 = K k0..k0+7,  halves 8-15 = K k0+16..k0+23
// lanes16-31: M=lane-16, same but K base +8 / +24.
__device__ __forceinline__ v16h load_a(const _Float16* base, int row, int ld,
                                       int k0, int lane) {
  int kb = k0 + ((lane & 16) ? 8 : 0);
  const _Float16* p = base + (size_t)row * ld + kb;
  union { uint4 u[2]; v16h h; } u;
  u.u[0] = *(const uint4*)p;
  u.u[1] = *(const uint4*)(p + 16);
  return u.h;
}

// B fragment (K x N), source stored row-major as W[n][k] (B column n = W row
// n, contiguous in k). Per SWMMAC 16-bit B layout halved:
// lanes 0-15: N=lane,    halves 0-7 = K k0..k0+7,  halves 8-15 = K k0+16..k0+23
// lanes16-31: N=lane-16, same but K base +8 / +24.
__device__ __forceinline__ v16h load_b(const _Float16* w, int nrow, int ld,
                                       int k0, int lane) {
  int kb = k0 + ((lane & 16) ? 8 : 0);
  const _Float16* p = w + (size_t)nrow * ld + kb;
  union { uint4 u[2]; v16h h; } u;
  u.u[0] = *(const uint4*)p;
  u.u[1] = *(const uint4*)(p + 16);
  return u.h;
}

// Cooperative global -> LDS copy (fallback staging path; ds_store_b128).
__device__ __forceinline__ void copy_to_lds(_Float16* dst,
                                            const _Float16* __restrict__ src,
                                            int nhalves, int tid, int nthr) {
  for (int i = tid * 8; i < nhalves; i += nthr * 8)
    *(uint4*)(dst + i) = *(const uint4*)(src + i);
}

#if NN48_HAVE_TDM
// Issue one 2-D TDM tile load: rows [0,tile_y) x cols [0,tile_x) elements of
// 2-byte data from `gptr` (row stride `stride_x` elements) into LDS offset
// `lds_off`, packed contiguously. D# layout per CDNA5 ISA 8.3/8.4.
__device__ __forceinline__ void tdm_load_2d(unsigned lds_off, const void* gptr,
                                            unsigned tile_x, unsigned tile_y,
                                            unsigned tensor_x, unsigned tensor_y,
                                            unsigned stride_x) {
  unsigned long long ga = (unsigned long long)gptr;
  v4u g0;
  g0[0] = 1u;                                   // count=1 (valid user D#)
  g0[1] = lds_off;                              // lds_addr (bytes)
  g0[2] = (unsigned)(ga & 0xFFFFFFFFu);         // global_addr[31:0]
  g0[3] = (unsigned)((ga >> 32) & 0x01FFFFFFu)  // global_addr[56:32]
          | (2u << 30);                         // type = 2 ("image")
  v8i g1;
  g1[0] = (int)(1u << 16);                      // wg_mask=0, data_size=1 (2B)
  g1[1] = (int)((tensor_x & 0xFFFFu) << 16);    // tensor_dim0[15:0] @ bit48
  g1[2] = (int)(((tensor_x >> 16) & 0xFFFFu) |
                ((tensor_y & 0xFFFFu) << 16));  // tensor_dim0 hi | tensor_dim1 lo
  g1[3] = (int)(((tensor_y >> 16) & 0xFFFFu) |
                ((tile_x & 0xFFFFu) << 16));    // tensor_dim1 hi | tile_dim0
  g1[4] = (int)(tile_y & 0xFFFFu);              // tile_dim1 | tile_dim2=0
  g1[5] = (int)stride_x;                        // tensor_dim0_stride[31:0]
  g1[6] = 0;                                    // stride0 hi | stride1 lo
  g1[7] = 0;                                    // stride1 hi
  v4i g2 = {0, 0, 0, 0};
  v4i g3 = {0, 0, 0, 0};
#if __clang_major__ >= 23
  v8i g4 = {0, 0, 0, 0, 0, 0, 0, 0};
  __builtin_amdgcn_tensor_load_to_lds(g0, g1, g2, g3, g4, 0);
#else
  __builtin_amdgcn_tensor_load_to_lds(g0, g1, g2, g3, 0);
#endif
}
#endif

// Stage a contiguous [ty x tx] f16 panel (row stride == tx) into LDS.
// TDM path: one tensor_load_to_lds issued by wave 0 (TDM ignores EXEC),
// wave 0 waits on TENSORcnt, then workgroup barrier publishes the panel.
__device__ __forceinline__ void stage_panel(_Float16* lds,
                                            const _Float16* __restrict__ src,
                                            unsigned tx, unsigned ty,
                                            int wave, int tid) {
#if NN48_HAVE_TDM
  if (wave == 0) {
    tdm_load_2d((unsigned)(unsigned long long)(uintptr_t)lds, src,
                tx, ty, tx, ty, tx);
    __builtin_amdgcn_s_wait_tensorcnt(0);
  }
#else
  copy_to_lds(lds, src, (int)(tx * ty), tid, 256);
#endif
  __syncthreads();
}

// ---- K0: weight prep (fold D into W12, f32->f16, pad W2) -------------------

__global__ void nn48_prep(const float* __restrict__ W11,
                          const float* __restrict__ W12,
                          const float* __restrict__ W2,
                          _Float16* __restrict__ w11h,
                          _Float16* __restrict__ w12ph,
                          _Float16* __restrict__ w2h) {
  using namespace cfg;
  const float c   = E_MOD / (1.0f - NU * NU);
  const float cnu = c * NU;
  const float cg  = c * (1.0f - NU) * 0.5f;
  const int stride = gridDim.x * blockDim.x;
  const int tid = blockIdx.x * blockDim.x + threadIdx.x;

  for (int i = tid; i < COH_SIZE * F; i += stride)
    w11h[i] = (_Float16)W11[i];

  for (int i = tid; i < BULK_SIZE * IN_BULK; i += stride) {
    int row = i / IN_BULK, k = i - row * IN_BULK;
    int p = row / 3, j = row - 3 * p;
    const float* w0 = W12 + (size_t)(p * 3) * IN_BULK;
    float v;
    if (j == 0)      v = c   * w0[k] + cnu * w0[IN_BULK + k];
    else if (j == 1) v = cnu * w0[k] + c   * w0[IN_BULK + k];
    else             v = cg  * w0[2 * IN_BULK + k];
    w12ph[i] = (_Float16)v;
  }

  for (int i = tid; i < 16 * BULK_SIZE; i += stride) {
    int row = i / BULK_SIZE, k = i - row * BULK_SIZE;
    w2h[i] = (row < OUT) ? (_Float16)W2[(size_t)row * BULK_SIZE + k]
                         : (_Float16)0.0f;
  }
}

// ---- K1: x f32 -> f16 ------------------------------------------------------

__global__ void nn48_x2h(const float* __restrict__ x, _Float16* __restrict__ xh,
                         int n4) {
  const int stride = gridDim.x * blockDim.x;
  for (int i = blockIdx.x * blockDim.x + threadIdx.x; i < n4; i += stride) {
    float4 f = ((const float4*)x)[i];
    v4h h;
    h[0] = (_Float16)f.x; h[1] = (_Float16)f.y;
    h[2] = (_Float16)f.z; h[3] = (_Float16)f.w;
    *(v4h*)(xh + (size_t)i * 4) = h;
  }
}

// ---- K2: GEMM1 (fc11) + fused leaky-relu + damage-onset d_new --------------
// Block: 256 threads = 8 waves; wave w -> m-tile (blockIdx.x*8 + w).
// W11 panel (128x128 f16 = 32 KB) staged in LDS via TDM, shared by 8 waves.
// B fragments software-pipelined one step ahead of the consuming WMMA.

__global__ void nn48_gemm1(const _Float16* __restrict__ xh,
                           const _Float16* __restrict__ w11h,
                           const float* __restrict__ b11,
                           float* __restrict__ dnew) {
  using namespace cfg;
  __shared__ _Float16 lds_w[COH_SIZE * F];     // 32 KB
  const int lane = threadIdx.x & 31;
  const int wave = threadIdx.x >> 5;
  stage_panel(lds_w, w11h, F, COH_SIZE, wave, threadIdx.x);

  const int m0 = (blockIdx.x * 8 + wave) * 16;
  const int nlo = lane & 15;
  const int mhalf = (lane >> 4) * 8;

  v8f acc[8] = {};
  v16h a_cur = load_a(xh, m0 + nlo, F, 0, lane);
  v16h b_cur = load_b(lds_w, nlo, F, 0, lane);
#pragma unroll
  for (int kk = 0; kk < 4; ++kk) {
    v16h a_nxt = a_cur;
    if (kk < 3) a_nxt = load_a(xh, m0 + nlo, F, (kk + 1) * 32, lane);
#pragma unroll
    for (int nt = 0; nt < 8; ++nt) {
      v16h b_nxt = b_cur;
      if (nt < 7)      b_nxt = load_b(lds_w, (nt + 1) * 16 + nlo, F, kk * 32, lane);
      else if (kk < 3) b_nxt = load_b(lds_w, nlo, F, (kk + 1) * 32, lane);
      acc[nt] = wmma16(a_cur, b_cur, acc[nt]);
      b_cur = b_nxt;
    }
    a_cur = a_nxt;
  }

  // Epilogue: +bias, leaky relu, pair (2c,2c+1)->d_new via lane shuffle.
#pragma unroll
  for (int nt = 0; nt < 8; ++nt) {
    const int n = nt * 16 + nlo;
    const float bias = b11[n];
#pragma unroll
    for (int r = 0; r < 8; ++r) {
      float j = acc[nt][r] + bias;
      j = (j > 0.0f) ? j : LEAKY * j;
      float jp = __shfl_xor(j, 1, 32);       // partner: opposite N parity
      if (!(nlo & 1)) {
        float jn = fmaxf(j, 0.0f);           // tension-only normal jump
        float js = jp;                        // shear jump
        float delta = sqrtf(jn * jn + js * js + EPS);
        float dn = DELTAF * (delta - DELTA0) /
                   (fmaxf(delta, EPS) * (DELTAF - DELTA0));
        dn = fminf(fmaxf(dn, 0.0f), 1.0f);
        dnew[(size_t)(m0 + mhalf + r) * COH + (n >> 1)] = dn;
      }
    }
  }
}

// ---- K3: cumulative max of d_new over T (damage never heals) ---------------

__global__ void nn48_cummax(const float* __restrict__ dnew,
                            _Float16* __restrict__ dmgh) {
  using namespace cfg;
  const int idx = blockIdx.x * blockDim.x + threadIdx.x;   // b*COH + c
  if (idx >= B * COH) return;
  const int b = idx / COH, c = idx - b * COH;
  float run = 0.0f;
  size_t off = (size_t)b * T * COH + c;
  for (int t = 0; t < T; ++t, off += COH) {
    run = fmaxf(run, dnew[off]);
    dmgh[off] = (_Float16)run;
  }
}

// ---- K4: GEMM2 (fc12 with D folded): sig_tr = [x|dmg] @ W12p^T -------------
// Grid (512, 3): each wave owns TWO m-tiles (32 rows) so every B fragment
// feeds two back-to-back WMMAs (independent accumulators, no D->A/B hazard),
// halving LDS traffic per MAC. B panel (192K x 128N = 48 KB) staged via TDM.
// K = 192: k-steps 0..3 read x_h (ld=F), 4..5 read dmg_h (ld=COH).

__global__ void nn48_gemm2(const _Float16* __restrict__ xh,
                           const _Float16* __restrict__ dmgh,
                           const _Float16* __restrict__ w12ph,
                           float* __restrict__ sig) {
  using namespace cfg;
  __shared__ _Float16 lds_b[128 * IN_BULK];    // 48 KB B panel
  const int lane = threadIdx.x & 31;
  const int wave = threadIdx.x >> 5;
  const int n0 = blockIdx.y * 128;
  stage_panel(lds_b, w12ph + (size_t)n0 * IN_BULK, IN_BULK, 128,
              wave, threadIdx.x);

  const int m0 = (blockIdx.x * 8 + wave) * 32;
  const int nlo = lane & 15;
  const int mhalf = (lane >> 4) * 8;

  auto a_at = [&](int row, int kk) {
    const _Float16* abase = (kk < 4) ? xh : dmgh;
    const int ld  = (kk < 4) ? F : COH;
    const int kof = (kk < 4) ? kk * 32 : (kk - 4) * 32;
    return load_a(abase, row, ld, kof, lane);
  };

  v8f acc0[8] = {}, acc1[8] = {};
  v16h a0_cur = a_at(m0 + nlo, 0);
  v16h a1_cur = a_at(m0 + 16 + nlo, 0);
  v16h b_cur  = load_b(lds_b, nlo, IN_BULK, 0, lane);
#pragma unroll
  for (int kk = 0; kk < 6; ++kk) {
    v16h a0_nxt = a0_cur, a1_nxt = a1_cur;
    if (kk < 5) {
      a0_nxt = a_at(m0 + nlo, kk + 1);
      a1_nxt = a_at(m0 + 16 + nlo, kk + 1);
    }
#pragma unroll
    for (int nt = 0; nt < 8; ++nt) {
      v16h b_nxt = b_cur;
      if (nt < 7)      b_nxt = load_b(lds_b, (nt + 1) * 16 + nlo, IN_BULK, kk * 32, lane);
      else if (kk < 5) b_nxt = load_b(lds_b, nlo, IN_BULK, (kk + 1) * 32, lane);
      acc0[nt] = wmma16(a0_cur, b_cur, acc0[nt]);
      acc1[nt] = wmma16(a1_cur, b_cur, acc1[nt]);
      b_cur = b_nxt;
    }
    a0_cur = a0_nxt;
    a1_cur = a1_nxt;
  }

#pragma unroll
  for (int nt = 0; nt < 8; ++nt)
#pragma unroll
    for (int r = 0; r < 8; ++r) {
      sig[(size_t)(m0 + mhalf + r) * BULK_SIZE + n0 + nt * 16 + nlo] =
          acc0[nt][r];
      sig[(size_t)(m0 + 16 + mhalf + r) * BULK_SIZE + n0 + nt * 16 + nlo] =
          acc1[nt][r];
    }
}

// ---- K5: J2 radial-return recurrence per (b, bulk_pt) chain ----------------
// ep_t = ep_{t-1} + relu(seq_t - (SY + H*ep_{t-1})) / (3G + H); scale stress.
// Writes the scaled stress as f16 (A-operand of GEMM3).

__global__ void nn48_plast(const float* __restrict__ sig,
                           _Float16* __restrict__ sigh) {
  using namespace cfg;
  const int idx = blockIdx.x * blockDim.x + threadIdx.x;   // b*BULK + p
  if (idx >= B * BULK) return;
  const int b = idx / BULK, p = idx - b * BULK;
  const float inv3GH = 1.0f / (3.0f * G_MOD + HARD);

  float ep = 0.0f;
  size_t off = (size_t)b * T * BULK_SIZE + (size_t)p * 3;
  for (int t = 0; t < T; ++t, off += BULK_SIZE) {
    float sxx = sig[off + 0];
    float syy = sig[off + 1];
    float txy = sig[off + 2];
    float seq = sqrtf(sxx * sxx - sxx * syy + syy * syy +
                      3.0f * txy * txy + EPS);
    float fy = seq - (SIG_Y + HARD * ep);
    float scale = 1.0f;
    if (fy > 0.0f) {
      ep += fy * inv3GH;
      scale = (SIG_Y + HARD * ep) / seq;
    }
    sigh[off + 0] = (_Float16)(sxx * scale);
    sigh[off + 1] = (_Float16)(syy * scale);
    sigh[off + 2] = (_Float16)(txy * scale);
  }
}

// ---- K6: GEMM3 (fc2, N padded 6->16) + fused softplus ----------------------

__global__ void nn48_gemm3(const _Float16* __restrict__ sigh,
                           const _Float16* __restrict__ w2h,
                           float* __restrict__ out) {
  using namespace cfg;
  __shared__ _Float16 lds_w2[16 * BULK_SIZE]; // 12 KB
  const int lane = threadIdx.x & 31;
  const int wave = threadIdx.x >> 5;
  stage_panel(lds_w2, w2h, BULK_SIZE, 16, wave, threadIdx.x);

  const int m0 = (blockIdx.x * 8 + wave) * 16;
  const int nlo = lane & 15;
  const int mhalf = (lane >> 4) * 8;

  v8f acc = {};
  v16h a_cur = load_a(sigh, m0 + nlo, BULK_SIZE, 0, lane);
  v16h b_cur = load_b(lds_w2, nlo, BULK_SIZE, 0, lane);
#pragma unroll
  for (int kk = 0; kk < 12; ++kk) {
    v16h a_nxt = a_cur, b_nxt = b_cur;
    if (kk < 11) {
      a_nxt = load_a(sigh, m0 + nlo, BULK_SIZE, (kk + 1) * 32, lane);
      b_nxt = load_b(lds_w2, nlo, BULK_SIZE, (kk + 1) * 32, lane);
    }
    acc = wmma16(a_cur, b_cur, acc);
    a_cur = a_nxt;
    b_cur = b_nxt;
  }

  if (nlo < OUT) {
#pragma unroll
    for (int r = 0; r < 8; ++r) {
      float v = acc[r];
      float sp = fmaxf(v, 0.0f) + log1pf(expf(-fabsf(v)));   // softplus
      out[(size_t)(m0 + mhalf + r) * OUT + nlo] = sp;
    }
  }
}

// ---------------------------------------------------------------------------

extern "C" void kernel_launch(void* const* d_in, const int* in_sizes, int n_in,
                              void* d_out, int out_size, void* d_ws,
                              size_t ws_size, hipStream_t stream) {
  using namespace cfg;
  const float* x   = (const float*)d_in[0];
  const float* W11 = (const float*)d_in[1];
  const float* b11 = (const float*)d_in[2];
  const float* W12 = (const float*)d_in[3];
  const float* W2  = (const float*)d_in[4];
  float* out = (float*)d_out;

  char* ws = (char*)d_ws;
  _Float16* w11h  = (_Float16*)(ws + wsl::o_w11h);
  _Float16* w12ph = (_Float16*)(ws + wsl::o_w12ph);
  _Float16* w2h   = (_Float16*)(ws + wsl::o_w2h);
  _Float16* xh    = (_Float16*)(ws + wsl::o_xh);
  float*    dnew  = (float*)   (ws + wsl::o_dnew);
  _Float16* dmgh  = (_Float16*)(ws + wsl::o_dmgh);
  float*    sig   = (float*)   (ws + wsl::o_sig);
  _Float16* sigh  = (_Float16*)(ws + wsl::o_sigh);

  const dim3 blk(256);

  nn48_prep<<<96, blk, 0, stream>>>(W11, W12, W2, w11h, w12ph, w2h);
  nn48_x2h<<<2048, blk, 0, stream>>>(x, xh, BT * F / 4);
  nn48_gemm1<<<BT / 128, blk, 0, stream>>>(xh, w11h, b11, dnew);
  nn48_cummax<<<(B * COH + 255) / 256, blk, 0, stream>>>(dnew, dmgh);
  nn48_gemm2<<<dim3(BT / 256, BULK_SIZE / 128), blk, 0, stream>>>(
      xh, dmgh, w12ph, sig);
  nn48_plast<<<(B * BULK + 255) / 256, blk, 0, stream>>>(sig, sigh);
  nn48_gemm3<<<BT / 128, blk, 0, stream>>>(sigh, w2h, out);
}
